// ConsensusNormalizingFlowDynamics_39633958207491
// MI455X (gfx1250) — matrix operand order repaired and checked by reading, not compiled, not run on hardware
//
#include <hip/hip_runtime.h>
#include <math.h>

// ---------------------------------------------------------------------------
// ConsensusNormalizingFlowDynamics on MI455X (gfx1250, wave32, WMMA f16)
// ---------------------------------------------------------------------------

typedef _Float16 v16h __attribute__((ext_vector_type(16)));
typedef __fp16   f16x2 __attribute__((ext_vector_type(2)));
typedef float    v8f  __attribute__((ext_vector_type(8)));

union V16U { v16h v; unsigned u[8]; };
union HU   { unsigned u; _Float16 h[2]; f16x2 v; };

#if defined(__AMDGCN__) && __has_builtin(__builtin_amdgcn_tanhf)
#define TANHF(x) __builtin_amdgcn_tanhf(x)
#elif defined(__AMDGCN__) && __has_builtin(__builtin_amdgcn_tanh_f32)
#define TANHF(x) __builtin_amdgcn_tanh_f32(x)
#else
#define TANHF(x) tanhf(x)
#endif

#define NETS 12
#define PFLT 4482           // floats per FCNN: w1 128, b1 64, w2 4096, b2 64, w3 128, b3 2
#define W2STRIDE 72         // halves per row (64 + 8 pad -> 36-bank lane stride, conflict free)
#define W2A_NET (64 * W2STRIDE)
#define W3A_NET (16 * W2STRIDE)

// LDS carve (bytes):
//  w2a 110592 + w3a 27648 + w14 12288 + b2 3072 + b3 96 + lap 1024
//  + km 256 + dm 256 + kphi 2048  = 157280
#define SMEM_BYTES 157280

static __device__ __forceinline__ v8f wmma16(v16h a, v16h b, v8f c) {
  return __builtin_amdgcn_wmma_f32_16x16x32_f16(false, a, false, b, (short)0, c,
                                                false, false);
}

static __device__ __forceinline__ unsigned pkh2(float a, float b) {
  HU t; t.v = __builtin_amdgcn_cvt_pkrtz(a, b); return t.u;
}

// Evaluate one 2->64->64->2 tanh MLP for 16 samples (one per lane pair) and its
// 2x2 input Jacobian via fused forward+tangent GEMMs.
//   h2^T = W2^T * [h1 | T0 | T1]^T   (A = W2^T from LDS, B built in regs)
//   out^T = W3pad^T * h2-variants    (C gives val/Jf directly on group-0 lanes)
static __device__ __forceinline__ void eval_fcnn(
    int net, int sRow, int g, float u0, float u1,
    const _Float16* __restrict__ w2a, const _Float16* __restrict__ w3a,
    const float4* __restrict__ w14, const float* __restrict__ b2f,
    const float* __restrict__ b3f,
    float& v0, float& v1, float (&Jf)[2][2])
{
  const float4* W14n = w14 + net * 64;      // (w1[0][k], w1[1][k], b1[k], 0)

  // ---- layer 1 (K=2) + tanh, build B fragments: Bh[tg][kf], K = 32kf+16g+t
  v16h Bh[3][2];
#pragma unroll
  for (int kf = 0; kf < 2; ++kf) {
#pragma unroll
    for (int t = 0; t < 16; ++t) {
      int k = 32 * kf + 16 * g + t;
      float4 q = W14n[k];
      float pre = __builtin_fmaf(u0, q.x, __builtin_fmaf(u1, q.y, q.z));
      float th = TANHF(pre);
      float dd = 1.0f - th * th;
      Bh[0][kf][t] = (_Float16)th;          // forward h1
      Bh[1][kf][t] = (_Float16)(dd * q.x);  // tangent col 0
      Bh[2][kf][t] = (_Float16)(dd * q.y);  // tangent col 1
    }
  }

  // ---- hidden GEMM: acc[mt][tg] = (W2^T tile mt) x Bh[tg]
  v8f acc[4][3];
  const _Float16* W2base = w2a + net * W2A_NET;
#pragma unroll
  for (int mt = 0; mt < 4; ++mt) {
    const _Float16* row = W2base + (sRow + 16 * mt) * W2STRIDE;
    V16U a0, a1;
#pragma unroll
    for (int p = 0; p < 8; ++p) {
      int kb = 2 * p + 16 * g;   // A-layout K pairs
      a0.u[p] = *(const unsigned*)(row + kb);
      a1.u[p] = *(const unsigned*)(row + 32 + kb);
    }
#pragma unroll
    for (int tg = 0; tg < 3; ++tg) {
      v8f z = {0.f, 0.f, 0.f, 0.f, 0.f, 0.f, 0.f, 0.f};
      z = wmma16(a0.v, Bh[tg][0], z);
      z = wmma16(a1.v, Bh[tg][1], z);
      acc[mt][tg] = z;
    }
  }

  // ---- bias + tanh + tangent scaling; rebuild as B fragments for W3 GEMM.
  // C layout: lane(s,g), tile mt, vgpr r  ->  n = 16*mt + 8*g + r.
  // B frag (kf): lane group g needs K=n in [32kf+16g, +15]; half comes from
  // the partner lane (xor 16).  Selects on g avoid dynamic register indexing.
  const float* B2p = b2f + net * 64;
  v16h Bo[3][2];
#pragma unroll
  for (int kf = 0; kf < 2; ++kf) {
    const int m0 = 2 * kf, m1 = 2 * kf + 1;
    const float4 cA0 = *(const float4*)(B2p + 16 * m0 + 8 * g);
    const float4 cA1 = *(const float4*)(B2p + 16 * m0 + 8 * g + 4);
    const float4 cB0 = *(const float4*)(B2p + 16 * m1 + 8 * g);
    const float4 cB1 = *(const float4*)(B2p + 16 * m1 + 8 * g + 4);
    const float b2a[8] = {cA0.x, cA0.y, cA0.z, cA0.w, cA1.x, cA1.y, cA1.z, cA1.w};
    const float b2b[8] = {cB0.x, cB0.y, cB0.z, cB0.w, cB1.x, cB1.y, cB1.z, cB1.w};
#pragma unroll
    for (int r = 0; r < 8; ++r) {
      float p0 = acc[m0][0][r] + b2a[r];
      float p1 = acc[m1][0][r] + b2b[r];
      float th0 = TANHF(p0), th1 = TANHF(p1);
      float thL  = g ? th1 : th0;                 // local (own K range)
      float thSd = g ? th0 : th1;                 // value partner needs
      float dL = 1.0f - thL * thL;
      float dS = 1.0f - thSd * thSd;
      float rcv0 = __shfl_xor(thSd, 16, 32);
      _Float16 eL0 = (_Float16)thL, eR0 = (_Float16)rcv0;
      Bo[0][kf][r]     = g ? eR0 : eL0;
      Bo[0][kf][8 + r] = g ? eL0 : eR0;
      // two tangent columns packed into one u32 per shuffle
      float tL0 = (g ? acc[m1][1][r] : acc[m0][1][r]) * dL;
      float tL1 = (g ? acc[m1][2][r] : acc[m0][2][r]) * dL;
      float tS0 = (g ? acc[m0][1][r] : acc[m1][1][r]) * dS;
      float tS1 = (g ? acc[m0][2][r] : acc[m1][2][r]) * dS;
      HU pl, pr;
      pl.u = pkh2(tL0, tL1);
      pr.u = __shfl_xor(pkh2(tS0, tS1), 16, 32);
      Bo[1][kf][r]     = g ? pr.h[0] : pl.h[0];
      Bo[2][kf][r]     = g ? pr.h[1] : pl.h[1];
      Bo[1][kf][8 + r] = g ? pl.h[0] : pr.h[0];
      Bo[2][kf][8 + r] = g ? pl.h[1] : pr.h[1];
    }
  }

  // ---- output GEMM: A = W3pad^T (rows 0,1 valid, rest zero)
  const _Float16* row3 = w3a + net * W3A_NET + sRow * W2STRIDE;
  V16U a20, a21;
#pragma unroll
  for (int p = 0; p < 8; ++p) {
    int kb = 2 * p + 16 * g;
    a20.u[p] = *(const unsigned*)(row3 + kb);
    a21.u[p] = *(const unsigned*)(row3 + 32 + kb);
  }
  v8f o[3];
#pragma unroll
  for (int tg = 0; tg < 3; ++tg) {
    v8f z = {0.f, 0.f, 0.f, 0.f, 0.f, 0.f, 0.f, 0.f};
    z = wmma16(a20.v, Bo[tg][0], z);
    z = wmma16(a21.v, Bo[tg][1], z);
    o[tg] = z;
  }
  // lane (s, g=0): VGPR r = output component j; group-1 lanes see zeros.
  v0 = o[0][0] + b3f[net * 2 + 0];
  v1 = o[0][1] + b3f[net * 2 + 1];
  Jf[0][0] = o[1][0]; Jf[1][0] = o[1][1];   // d out_j / d u0
  Jf[0][1] = o[2][0]; Jf[1][1] = o[2][1];   // d out_j / d u1
}

// Coupling-layer update: p_i' = vt_i + p_i * exp(vs_i); chain rule on (Jp,Ja).
static __device__ __forceinline__ void pair_update(
    float vs0, float vs1, const float (&Js)[2][2],
    float vt0, float vt1, const float (&Jt)[2][2],
    float& p0, float& p1, float (&Jp)[2][4], const float (&Ja)[2][4])
{
  float e0 = __expf(vs0), e1 = __expf(vs1);
  float np0 = __builtin_fmaf(p0, e0, vt0);
  float np1 = __builtin_fmaf(p1, e1, vt1);
#pragma unroll
  for (int k = 0; k < 4; ++k) {
    float As0 = Js[0][0] * Ja[0][k] + Js[0][1] * Ja[1][k];
    float As1 = Js[1][0] * Ja[0][k] + Js[1][1] * Ja[1][k];
    float At0 = Jt[0][0] * Ja[0][k] + Jt[0][1] * Ja[1][k];
    float At1 = Jt[1][0] * Ja[0][k] + Jt[1][1] * Ja[1][k];
    Jp[0][k] = At0 + e0 * Jp[0][k] + p0 * e0 * As0;
    Jp[1][k] = At1 + e1 * Jp[1][k] + p1 * e1 * As1;
  }
  p0 = np0; p1 = np1;
}

__global__ __launch_bounds__(256, 1)
void cnf_dyn_kernel(const float* __restrict__ x, const float* __restrict__ xd,
                    const float* __restrict__ prm, const float* __restrict__ lap,
                    const float* __restrict__ km, const float* __restrict__ dm,
                    float* __restrict__ out)
{
  extern __shared__ char smem_raw[];
  _Float16* w2a = reinterpret_cast<_Float16*>(smem_raw);
  _Float16* w3a = w2a + NETS * W2A_NET;
  float4* w14 = reinterpret_cast<float4*>(w3a + NETS * W3A_NET);
  float* b2f  = reinterpret_cast<float*>(w14 + NETS * 64);
  float* b3f  = b2f + NETS * 64;
  float* lapS = b3f + NETS * 2;
  float* kmS  = lapS + 256;
  float* dmS  = kmS + 64;
  float* kphiS = dmS + 64;                  // 128 samples x 4

  const int tid = threadIdx.x;

  // ---------------- cooperative param pack (f32 -> f16, transposed) ---------
#pragma unroll 1
  for (int net = 0; net < NETS; ++net) {
    const float* pb = prm + net * PFLT;
    _Float16* W2d = w2a + net * W2A_NET;
    for (int i = tid; i < 4096; i += 256) {
      int k = i >> 6, n = i & 63;
      W2d[n * W2STRIDE + k] = (_Float16)pb[192 + k * 64 + n];    // W2^T
    }
    _Float16* W3d = w3a + net * W3A_NET;
    for (int i = tid; i < 1024; i += 256) {
      int k = i >> 4, m = i & 15;
      W3d[m * W2STRIDE + k] = (_Float16)((m < 2) ? pb[4352 + k * 2 + m] : 0.0f);
    }
    if (tid < 64) {
      float4 q;
      q.x = pb[tid];            // w1[0][k]
      q.y = pb[64 + tid];       // w1[1][k]
      q.z = pb[128 + tid];      // b1[k]
      q.w = 0.0f;
      w14[net * 64 + tid] = q;
      b2f[net * 64 + tid] = pb[4288 + tid];
    }
    if (tid < 2) b3f[net * 2 + tid] = pb[4480 + tid];
  }
  lapS[tid] = lap[tid];
  if (tid < 64)  kmS[tid] = km[tid];
  if (tid >= 64 && tid < 128) dmS[tid - 64] = dm[tid - 64];
  __syncthreads();

  // ---------------- per-wave sample setup ----------------------------------
  const int lane = tid & 31;
  const int wv   = tid >> 5;
  const int sRow = lane & 15;               // sample within wave / A-frag row
  const int g    = lane >> 4;               // lane half-group
  const int samp = blockIdx.x * 128 + wv * 16 + sRow;
  const int bIx  = samp >> 2;               // batch item
  const int aIx  = samp & 3;                // agent

  // z0 = (x @ lap^T) row for this (b, a)
  const float* xrow = x + (size_t)bIx * 16;
  float z0[4];
#pragma unroll
  for (int i = 0; i < 4; ++i) {
    float accv = 0.f;
    const float* lr = lapS + (aIx * 4 + i) * 16;
#pragma unroll
    for (int j = 0; j < 16; ++j) accv = __builtin_fmaf(lr[j], xrow[j], accv);
    z0[i] = accv;
  }
  float low0 = z0[0], low1 = z0[1], up0 = z0[2], up1 = z0[3];
  float JL[2][4] = {{1.f, 0.f, 0.f, 0.f}, {0.f, 1.f, 0.f, 0.f}};
  float JU[2][4] = {{0.f, 0.f, 1.f, 0.f}, {0.f, 0.f, 0.f, 1.f}};

  // ---------------- 3 RealNVP flows ----------------------------------------
#pragma unroll 1
  for (int flow = 0; flow < 3; ++flow) {
    const int ns1 = flow * 4 + 0, ns2 = flow * 4 + 1;   // alphabetical: s1,s2,t1,t2
    const int nt1 = flow * 4 + 2, nt2 = flow * 4 + 3;
    float vs0, vs1, vt0, vt1;
    float Js[2][2], Jt[2][2];

    // pair 1: upper <- t1(lower) + upper * exp(s1(lower))
    float u0 = __shfl(low0, sRow, 32);
    float u1 = __shfl(low1, sRow, 32);
    eval_fcnn(ns1, sRow, g, u0, u1, w2a, w3a, w14, b2f, b3f, vs0, vs1, Js);
    eval_fcnn(nt1, sRow, g, u0, u1, w2a, w3a, w14, b2f, b3f, vt0, vt1, Jt);
    pair_update(vs0, vs1, Js, vt0, vt1, Jt, up0, up1, JU, JL);

    // pair 2: lower <- t2(upper') + lower * exp(s2(upper'))
    u0 = __shfl(up0, sRow, 32);
    u1 = __shfl(up1, sRow, 32);
    eval_fcnn(ns2, sRow, g, u0, u1, w2a, w3a, w14, b2f, b3f, vs0, vs1, Js);
    eval_fcnn(nt2, sRow, g, u0, u1, w2a, w3a, w14, b2f, b3f, vt0, vt1, Jt);
    pair_update(vs0, vs1, Js, vt0, vt1, Jt, low0, low1, JL, JU);
  }

  // ---------------- combine: spring + damping ------------------------------
  float ph[4] = {low0, low1, up0, up1};
  float J4[4][4];
#pragma unroll
  for (int k = 0; k < 4; ++k) {
    J4[0][k] = JL[0][k]; J4[1][k] = JL[1][k];
    J4[2][k] = JU[0][k]; J4[3][k] = JU[1][k];
  }

  // Kphi (per agent) -> LDS for cross-agent laplacian coupling
#pragma unroll
  for (int i = 0; i < 4; ++i) {
    float accv = 0.f;
#pragma unroll
    for (int j = 0; j < 4; ++j)
      accv = __builtin_fmaf(kmS[aIx * 16 + i * 4 + j], ph[j], accv);
    if (lane < 16) kphiS[(wv * 16 + sRow) * 4 + i] = accv;
  }
  __syncthreads();

  const int grp = wv * 16 + (sRow & ~3);    // 4 agents of this batch item
  float lk[4], us[4];
#pragma unroll
  for (int i = 0; i < 4; ++i) {
    float accv = 0.f;
    const float* lr = lapS + (aIx * 4 + i) * 16;
#pragma unroll
    for (int j = 0; j < 16; ++j)
      accv = __builtin_fmaf(lr[j], kphiS[(grp + (j >> 2)) * 4 + (j & 3)], accv);
    lk[i] = accv;
  }
#pragma unroll
  for (int i = 0; i < 4; ++i) {             // u_spring = J^T LKphi
    us[i] = J4[0][i] * lk[0] + J4[1][i] * lk[1] + J4[2][i] * lk[2] + J4[3][i] * lk[3];
  }

  const float* xdp = xd + (size_t)samp * 4;
  float xv[4] = {xdp[0], xdp[1], xdp[2], xdp[3]};
  float jx[4], dj[4];
#pragma unroll
  for (int i = 0; i < 4; ++i)               // Jxd
    jx[i] = J4[i][0] * xv[0] + J4[i][1] * xv[1] + J4[i][2] * xv[2] + J4[i][3] * xv[3];
#pragma unroll
  for (int i = 0; i < 4; ++i) {             // D Jxd
    dj[i] = dmS[aIx * 16 + i * 4 + 0] * jx[0] + dmS[aIx * 16 + i * 4 + 1] * jx[1] +
            dmS[aIx * 16 + i * 4 + 2] * jx[2] + dmS[aIx * 16 + i * 4 + 3] * jx[3];
  }
  if (lane < 16) {
    float4 r;
    r.x = us[0] - (J4[0][0] * dj[0] + J4[1][0] * dj[1] + J4[2][0] * dj[2] + J4[3][0] * dj[3]);
    r.y = us[1] - (J4[0][1] * dj[0] + J4[1][1] * dj[1] + J4[2][1] * dj[2] + J4[3][1] * dj[3]);
    r.z = us[2] - (J4[0][2] * dj[0] + J4[1][2] * dj[1] + J4[2][2] * dj[2] + J4[3][2] * dj[3]);
    r.w = us[3] - (J4[0][3] * dj[0] + J4[1][3] * dj[1] + J4[2][3] * dj[2] + J4[3][3] * dj[3]);
    *reinterpret_cast<float4*>(out + (size_t)samp * 4) = r;
  }
}

// ---------------------------------------------------------------------------
extern "C" void kernel_launch(void* const* d_in, const int* in_sizes, int n_in,
                              void* d_out, int out_size, void* d_ws, size_t ws_size,
                              hipStream_t stream) {
  const float* x  = (const float*)d_in[0];
  const float* xd = (const float*)d_in[1];
  const float *prm, *lap, *km, *dm;

  if (n_in >= 75) {
    // flow_params passed as 72 separate leaves (pytree order: flows outer,
    // dict keys alphabetical s1,s2,t1,t2, each (w1,b1,w2,b2,w3,b3)) -> pack
    // into workspace in the canonical contiguous layout.
    static const int leaf[6] = {128, 64, 4096, 64, 128, 2};
    float* pk = (float*)d_ws;
    size_t off = 0;
    int idx = 2;
    for (int net = 0; net < NETS; ++net)
      for (int l = 0; l < 6; ++l) {
        (void)hipMemcpyAsync(pk + off, d_in[idx], (size_t)leaf[l] * sizeof(float),
                             hipMemcpyDeviceToDevice, stream);
        off += leaf[l]; ++idx;
      }
    prm = pk;
    lap = (const float*)d_in[n_in - 3];
    km  = (const float*)d_in[n_in - 2];
    dm  = (const float*)d_in[n_in - 1];
  } else {
    prm = (const float*)d_in[2];
    lap = (const float*)d_in[3];
    km  = (const float*)d_in[4];
    dm  = (const float*)d_in[5];
  }

  const int batchB = in_sizes[0] / 16;          // BATCH
  const int blocks = (batchB * 4) / 128;        // 16 samples/wave, 8 waves/block

  (void)hipFuncSetAttribute((const void*)cnf_dyn_kernel,
                            hipFuncAttributeMaxDynamicSharedMemorySize, SMEM_BYTES);
  cnf_dyn_kernel<<<blocks, 256, SMEM_BYTES, stream>>>(x, xd, prm, lap, km, dm,
                                                      (float*)d_out);
}
